// NCTR_60498909331816
// MI455X (gfx1250) — matrix operand order repaired and checked by reading, not compiled, not run on hardware
//
#include <hip/hip_runtime.h>
#include <hip/hip_bf16.h>
#include <math.h>

// ---------------------------------------------------------------------------
// Types for CDNA5 WMMA (wave32): D(f32 16x16) = A(bf16 16x32) x B(bf16 32x16) + C
// ---------------------------------------------------------------------------
typedef __bf16 bf;
typedef __attribute__((ext_vector_type(16))) __bf16 v16bf;
typedef __attribute__((ext_vector_type(8)))  __bf16 v8bf;
typedef __attribute__((ext_vector_type(8)))  float  v8f;

union VA16 { v16bf v; v8bf h[2]; };
union PU16 { v16bf v; bf e[16]; };

static __device__ inline v8f wmma_bf16(v16bf a, v16bf b, v8f c) {
    // 8 args: (neg_a, A, neg_b, B, c_mod, C, reuse_a, reuse_b)
    return __builtin_amdgcn_wmma_f32_16x16x32_bf16(false, a, false, b, (short)0, c,
                                                   false, false);
}

// A operand (16x32, M x K), row-major source with leading dim lda.
// Lane L<16 : row m0+L, K = {k0+0..7, k0+16..23}
// Lane L>=16: row m0+(L-16), K = {k0+8..15, k0+24..31}
static __device__ inline v16bf load_a16(const bf* A, int lda, int m0, int k0, int lane) {
    const int hi = lane >> 4;
    const bf* p = A + (size_t)(m0 + (lane & 15)) * lda + k0 + hi * 8;
    VA16 u;
    u.h[0] = *(const v8bf*)p;
    u.h[1] = *(const v8bf*)(p + 16);
    return u.v;
}

// B operand (32x16, K x N) where B[k][n] = W[n][k], W row-major with ldw.
// Lane L<16 : col n0+L, K = {k0+0..7, k0+16..23} ; lane L>=16: col n0+(L-16), K = {k0+8..15, k0+24..31}
static __device__ inline v16bf load_b16(const bf* W, int ldw, int n0, int k0, int lane) {
    const int hi = lane >> 4;
    const bf* p = W + (size_t)(n0 + (lane & 15)) * ldw + k0 + hi * 8;
    VA16 u;
    u.h[0] = *(const v8bf*)p;
    u.h[1] = *(const v8bf*)(p + 16);
    return u.v;
}

// ---------------------------------------------------------------------------
// Problem constants (fixed by reference setup)
// ---------------------------------------------------------------------------
#define BZ   4
#define NTOK 2048
#define DM   256
#define ROWS (BZ * NTOK)     // 8192
#define SKN  2049            // Sinkhorn side (n+1)
#define NORM_C  (-8.31776617f)   // -log(4096)
#define LOGN_C  ( 7.62461899f)   //  log(2048)
#define CSLAB 16

// ---------------------------------------------------------------------------
// Generic bf16 WMMA GEMM: C[m][n] = epi(scale * sum_k A[m,k]*W[n,k] + bias[n])
// block = 256 threads = 8 waves -> tile 128(M) x 64(N), wave tile 32x32.
// M,N,K must be multiples of 128/64/32 (true for all uses here).
// ---------------------------------------------------------------------------
template<bool HAS_BIAS, bool DO_GELU, bool OUTF, bool OUTB>
__global__ __launch_bounds__(256) void gemm_bf16_k(
        const bf* __restrict__ A, int lda, long long sA,
        const bf* __restrict__ W, int ldw, long long sW,
        const float* __restrict__ bias,
        float* __restrict__ Cf, bf* __restrict__ Cb, int ldc, long long sC,
        int K, float scale)
{
    const int bz = blockIdx.z;
    A += (size_t)bz * sA;
    W += (size_t)bz * sW;
    const int lane = threadIdx.x & 31;
    const int wave = threadIdx.x >> 5;
    const int hi = lane >> 4, ln = lane & 15;
    const int m0 = blockIdx.y * 128 + (wave >> 1) * 32;
    const int n0 = blockIdx.x * 64  + (wave & 1) * 32;

    v8f acc00 = {}, acc01 = {}, acc10 = {}, acc11 = {};
#pragma unroll 2
    for (int k0 = 0; k0 < K; k0 += 32) {
        v16bf a0 = load_a16(A, lda, m0,      k0, lane);
        v16bf a1 = load_a16(A, lda, m0 + 16, k0, lane);
        v16bf b0 = load_b16(W, ldw, n0,      k0, lane);
        v16bf b1 = load_b16(W, ldw, n0 + 16, k0, lane);
        acc00 = wmma_bf16(a0, b0, acc00);
        acc01 = wmma_bf16(a0, b1, acc01);
        acc10 = wmma_bf16(a1, b0, acc10);
        acc11 = wmma_bf16(a1, b1, acc11);
    }

    float* cf = Cf + (size_t)bz * sC;
    bf*    cb = Cb + (size_t)bz * sC;
    auto emit = [&](float v, int row, int col) {
        v *= scale;
        if (HAS_BIAS) v += bias[col];
        if (DO_GELU)  v = 0.5f * v * (1.0f + erff(v * 0.70710678f));
        if (OUTF) cf[(size_t)row * ldc + col] = v;
        if (OUTB) cb[(size_t)row * ldc + col] = (bf)v;
    };
#pragma unroll
    for (int r = 0; r < 8; ++r) {
        const int row = m0 + r + 8 * hi;
        emit(acc00[r], row,      n0 + ln);
        emit(acc01[r], row,      n0 + 16 + ln);
        emit(acc10[r], row + 16, n0 + ln);
        emit(acc11[r], row + 16, n0 + 16 + ln);
    }
}

// ---------------------------------------------------------------------------
// Transpose + LayerNorm: X [B, 256, 2048] f32 -> out bf16 rows [B*2048, ldo]
// (LN over the 256-channel axis).  32 tokens per block via LDS transpose.
// ---------------------------------------------------------------------------
__global__ __launch_bounds__(256) void trans_ln_k(
        const float* __restrict__ X, const float* __restrict__ g,
        const float* __restrict__ h, bf* __restrict__ out, int ldo)
{
    const int b = blockIdx.y, n0 = blockIdx.x * 32;
    __shared__ float tile[256][33];
    __shared__ float red[2][8][32];
    __shared__ float smu[32], srs[32];
    const int t = threadIdx.x, x = t & 31, y = t >> 5;

    for (int d = y; d < 256; d += 8)
        tile[d][x] = X[((size_t)b * 256 + d) * NTOK + n0 + x];
    __syncthreads();

    float s = 0.f, q = 0.f;
    for (int d = y * 32; d < y * 32 + 32; ++d) {
        float v = tile[d][x];
        s += v; q += v * v;
    }
    red[0][y][x] = s; red[1][y][x] = q;
    __syncthreads();
    if (y == 0) {
        float S = 0.f, Q = 0.f;
        for (int k = 0; k < 8; ++k) { S += red[0][k][x]; Q += red[1][k][x]; }
        float mu = S * (1.0f / 256.0f);
        float var = Q * (1.0f / 256.0f) - mu * mu;
        smu[x] = mu; srs[x] = rsqrtf(var + 1e-5f);
    }
    __syncthreads();
    const float gg = g[t], hh = h[t];
    for (int c = 0; c < 32; ++c) {
        float v = (tile[t][c] - smu[c]) * srs[c] * gg + hh;
        out[((size_t)b * NTOK + n0 + c) * ldo + t] = (bf)v;
    }
}

// Row LayerNorm over 256 channels: in f32 [R,256]; optional f32 / bf16 outputs.
__global__ __launch_bounds__(256) void row_ln_k(
        const float* __restrict__ X, const float* __restrict__ g,
        const float* __restrict__ h, float* __restrict__ outF,
        bf* __restrict__ outB, int ldo)
{
    const int r = blockIdx.x, t = threadIdx.x;
    __shared__ float s1[256], s2[256];
    float v = X[(size_t)r * 256 + t];
    s1[t] = v; s2[t] = v * v;
    __syncthreads();
    for (int off = 128; off > 0; off >>= 1) {
        if (t < off) { s1[t] += s1[t + off]; s2[t] += s2[t + off]; }
        __syncthreads();
    }
    float mu = s1[0] * (1.0f / 256.0f);
    float var = s2[0] * (1.0f / 256.0f) - mu * mu;
    float o = (v - mu) * rsqrtf(var + 1e-5f) * g[t] + h[t];
    if (outF) outF[(size_t)r * 256 + t] = o;
    if (outB) outB[(size_t)r * ldo + t] = (bf)o;
}

// descnew_nT[b,n,c] = desc[b,c,n] + delta[b,n,c]  (bf16 out, LDS transpose)
__global__ __launch_bounds__(256) void trans_add_k(
        const float* __restrict__ desc, const float* __restrict__ delta,
        bf* __restrict__ out)
{
    const int b = blockIdx.y, n0 = blockIdx.x * 32;
    __shared__ float tile[256][33];
    const int t = threadIdx.x, x = t & 31, y = t >> 5;
    for (int d = y; d < 256; d += 8)
        tile[d][x] = desc[((size_t)b * 256 + d) * NTOK + n0 + x];
    __syncthreads();
    for (int c = 0; c < 32; ++c) {
        size_t rr = (size_t)b * NTOK + n0 + c;
        out[rr * 256 + t] = (bf)(tile[t][c] + delta[rr * 256 + t]);
    }
}

// ---------------------------------------------------------------------------
// Flash attention, per (q-block, head, batch).  8 waves x 16 q-rows = 128 rows.
// Q,K,V,O: bf16 [B, 2048, 256]; head h occupies columns [64h, 64h+64).
// ---------------------------------------------------------------------------
__global__ __launch_bounds__(256) void flash_attn_k(
        const bf* __restrict__ Q, const bf* __restrict__ K,
        const bf* __restrict__ V, bf* __restrict__ O)
{
    const int b = blockIdx.z, hh = blockIdx.y;
    const int tid = threadIdx.x;
    const int lane = tid & 31, wave = tid >> 5;
    const int hi = lane >> 4, ln = lane & 15;
    const int qrow0 = blockIdx.x * 128 + wave * 16;
    const size_t hoff = (size_t)b * NTOK * 256 + hh * 64;
    const bf* Qb = Q + hoff; const bf* Kb = K + hoff; const bf* Vb = V + hoff;

    __shared__ __align__(16) bf sK[32][64];     // K chunk, row-major (m, c)
    __shared__ __align__(16) bf sVt[64][40];    // V chunk transposed (d, m), padded
    __shared__ float sS[8][16][33];             // per-wave scores / probs
    __shared__ float sCorr[8][16];

    const v16bf qa0 = load_a16(Qb, 256, qrow0, 0,  lane);
    const v16bf qa1 = load_a16(Qb, 256, qrow0, 32, lane);

    v8f o0 = {}, o1 = {}, o2 = {}, o3 = {};
    float runM = -INFINITY, runS = 0.f;

    for (int mc = 0; mc < NTOK; mc += 32) {
        __syncthreads();
        {   // cooperative stage of K (row-major) and V (transposed) chunks
            const int idx = tid * 8;
            const int r = idx >> 6, c = idx & 63;
            *(v8bf*)&sK[r][c] = *(const v8bf*)(Kb + (size_t)(mc + r) * 256 + c);
            v8bf vv = *(const v8bf*)(Vb + (size_t)(mc + r) * 256 + c);
#pragma unroll
            for (int j = 0; j < 8; ++j) sVt[c + j][r] = vv[j];
        }
        __syncthreads();

        // S = Q * K^T (16 x 32), K-dim = 64 over two WMMA steps per m-subtile
        v8f s0 = {}, s1 = {};
        s0 = wmma_bf16(qa0, load_b16(&sK[0][0], 64, 0,  0,  lane), s0);
        s0 = wmma_bf16(qa1, load_b16(&sK[0][0], 64, 0,  32, lane), s0);
        s1 = wmma_bf16(qa0, load_b16(&sK[0][0], 64, 16, 0,  lane), s1);
        s1 = wmma_bf16(qa1, load_b16(&sK[0][0], 64, 16, 32, lane), s1);
#pragma unroll
        for (int r = 0; r < 8; ++r) {
            sS[wave][r + 8 * hi][ln]      = s0[r] * 0.125f;   // 1/sqrt(64)
            sS[wave][r + 8 * hi][16 + ln] = s1[r] * 0.125f;
        }
        __syncthreads();

        // online softmax: lanes 0..15 each own one q-row
        if (lane < 16) {
            float cm = -INFINITY;
#pragma unroll
            for (int j = 0; j < 32; ++j) cm = fmaxf(cm, sS[wave][lane][j]);
            const float newM = fmaxf(runM, cm);
            const float corr = __expf(runM - newM);
            float ps = 0.f;
#pragma unroll
            for (int j = 0; j < 32; ++j) {
                float p = __expf(sS[wave][lane][j] - newM);
                sS[wave][lane][j] = p;
                ps += p;
            }
            runS = runS * corr + ps;
            runM = newM;
            sCorr[wave][lane] = corr;
        }
        __syncthreads();

#pragma unroll
        for (int r = 0; r < 8; ++r) {
            const float c = sCorr[wave][r + 8 * hi];
            o0[r] *= c; o1[r] *= c; o2[r] *= c; o3[r] *= c;
        }
        // probs -> WMMA A operand (16x32), per-lane K map per ISA layout
        PU16 pu;
#pragma unroll
        for (int j = 0; j < 16; ++j) {
            const int kk = ((j < 8) ? j : j + 8) + hi * 8;
            pu.e[j] = (bf)sS[wave][ln][kk];
        }
        o0 = wmma_bf16(pu.v, load_b16(&sVt[0][0], 40, 0,  0, lane), o0);
        o1 = wmma_bf16(pu.v, load_b16(&sVt[0][0], 40, 16, 0, lane), o1);
        o2 = wmma_bf16(pu.v, load_b16(&sVt[0][0], 40, 32, 0, lane), o2);
        o3 = wmma_bf16(pu.v, load_b16(&sVt[0][0], 40, 48, 0, lane), o3);
    }

    __syncthreads();
    if (lane < 16) sCorr[wave][lane] = 1.0f / runS;
    __syncthreads();
    bf* Ob = O + hoff;
#pragma unroll
    for (int r = 0; r < 8; ++r) {
        const float inv = sCorr[wave][r + 8 * hi];
        const size_t row = (size_t)(qrow0 + r + 8 * hi) * 256;
        Ob[row + 0  + ln] = (bf)(o0[r] * inv);
        Ob[row + 16 + ln] = (bf)(o1[r] * inv);
        Ob[row + 32 + ln] = (bf)(o2[r] * inv);
        Ob[row + 48 + ln] = (bf)(o3[r] * inv);
    }
}

// ---------------------------------------------------------------------------
// Small utility kernels
// ---------------------------------------------------------------------------
__global__ void f2bf_k(const float* __restrict__ in, bf* __restrict__ out, int n) {
    const int i = blockIdx.x * 256 + threadIdx.x;
    if (i < n) out[i] = (bf)in[i];
}
__global__ void zero_k(float* __restrict__ p, int n) {
    const int i = blockIdx.x * 256 + threadIdx.x;
    if (i < n) p[i] = 0.f;
}
__global__ void fill_bins_k(float* __restrict__ coup, const float* __restrict__ alpha) {
    const float a = *alpha;
    const int i = blockIdx.x * 256 + threadIdx.x;
    if (i >= BZ * SKN) return;
    const int b = i / SKN, j = i % SKN;
    float* C = coup + (size_t)b * SKN * SKN;
    C[(size_t)j * SKN + (SKN - 1)] = a;         // last column
    C[(size_t)(SKN - 1) * SKN + j] = a;         // last row (corner consistent)
}

// ---------------------------------------------------------------------------
// Sinkhorn (log-space).  u[b,i] = log_mu[i] - LSE_j(coup[b,i,j] + v[b,j])
// ---------------------------------------------------------------------------
static __device__ inline void lse_update(float& m, float& s, float c) {
    if (c > m) { s = s * __expf(m - c) + 1.0f; m = c; }
    else       { s += __expf(c - m); }
}

__global__ __launch_bounds__(256) void sink_row_k(
        const float* __restrict__ coup, const float* __restrict__ v,
        float* __restrict__ u)
{
    const int b = blockIdx.y, i = blockIdx.x, t = threadIdx.x;
    const float* row = coup + ((size_t)b * SKN + i) * SKN;
    const float* vv = v + (size_t)b * SKN;
    float m = -INFINITY, s = 0.f;
    for (int j = t; j < SKN; j += 256) lse_update(m, s, row[j] + vv[j]);
    __shared__ float sm[256], ss[256];
    sm[t] = m; ss[t] = s;
    __syncthreads();
    for (int off = 128; off > 0; off >>= 1) {
        if (t < off) {
            float m2 = sm[t + off], s2 = ss[t + off];
            float M = fmaxf(sm[t], m2);
            ss[t] = ss[t] * __expf(sm[t] - M) + s2 * __expf(m2 - M);
            sm[t] = M;
        }
        __syncthreads();
    }
    if (t == 0) {
        const float lse = sm[0] + __logf(ss[0]);
        const float lmu = (i < SKN - 1) ? NORM_C : (LOGN_C + NORM_C);
        u[(size_t)b * SKN + i] = lmu - lse;
    }
}

// column pass, sliced into CSLAB i-slabs for parallelism; partial (m,s)
__global__ __launch_bounds__(256) void sink_col_part_k(
        const float* __restrict__ coup, const float* __restrict__ u,
        float* __restrict__ pm, float* __restrict__ ps)
{
    const int b = blockIdx.y, slab = blockIdx.z;
    const int j = blockIdx.x * 256 + threadIdx.x;
    if (j >= SKN) return;
    const int i0 = slab * 129;
    const int i1 = (i0 + 129 < SKN) ? i0 + 129 : SKN;
    const float* C = coup + (size_t)b * SKN * SKN;
    const float* uu = u + (size_t)b * SKN;
    float m = -INFINITY, s = 0.f;
    for (int i = i0; i < i1; ++i) {
        // hide GL2 hit latency of the 2049-stride column walk (branchless clamp)
        const int ipf = (i + 6 < i1) ? (i + 6) : i;
        __builtin_prefetch(&C[(size_t)ipf * SKN + j], 0, 3);
        lse_update(m, s, C[(size_t)i * SKN + j] + uu[i]);
    }
    pm[((size_t)b * CSLAB + slab) * SKN + j] = m;
    ps[((size_t)b * CSLAB + slab) * SKN + j] = s;
}

__global__ __launch_bounds__(256) void sink_col_comb_k(
        const float* __restrict__ pm, const float* __restrict__ ps,
        float* __restrict__ v)
{
    const int b = blockIdx.y;
    const int j = blockIdx.x * 256 + threadIdx.x;
    if (j >= SKN) return;
    float m = -INFINITY, s = 0.f;
    for (int k = 0; k < CSLAB; ++k) {
        const float m2 = pm[((size_t)b * CSLAB + k) * SKN + j];
        const float s2 = ps[((size_t)b * CSLAB + k) * SKN + j];
        const float M = fmaxf(m, m2);
        s = s * __expf(m - M) + s2 * __expf(m2 - M);
        m = M;
    }
    const float lnu = (j < SKN - 1) ? NORM_C : (LOGN_C + NORM_C);
    v[(size_t)b * SKN + j] = lnu - (m + __logf(s));
}

__global__ void sink_final_k(const float* __restrict__ coup,
                             const float* __restrict__ u,
                             const float* __restrict__ v,
                             float* __restrict__ out)
{
    const size_t idx = (size_t)blockIdx.x * 256 + threadIdx.x;
    const size_t total = (size_t)BZ * SKN * SKN;
    if (idx >= total) return;
    const size_t per = (size_t)SKN * SKN;
    const int b = (int)(idx / per);
    const size_t rem = idx % per;
    const int i = (int)(rem / SKN), j = (int)(rem % SKN);
    out[idx] = coup[idx] + u[(size_t)b * SKN + i] + v[(size_t)b * SKN + j] - NORM_C;
}

// ---------------------------------------------------------------------------
// Host orchestration
// ---------------------------------------------------------------------------
extern "C" void kernel_launch(void* const* d_in, const int* in_sizes, int n_in,
                              void* d_out, int out_size, void* d_ws, size_t ws_size,
                              hipStream_t stream)
{
    (void)in_sizes; (void)n_in; (void)out_size; (void)ws_size;
    const float* desc0 = (const float*)d_in[0];
    const float* desc1 = (const float*)d_in[1];
    const float* wq = (const float*)d_in[2];  const float* bq = (const float*)d_in[3];
    const float* wk = (const float*)d_in[4];  const float* bk = (const float*)d_in[5];
    const float* wv = (const float*)d_in[6];  const float* bv = (const float*)d_in[7];
    const float* wm = (const float*)d_in[8];  const float* bm = (const float*)d_in[9];
    const float* w1 = (const float*)d_in[10]; const float* c1 = (const float*)d_in[11];
    const float* w2 = (const float*)d_in[12]; const float* c2 = (const float*)d_in[13];
    const float* g0 = (const float*)d_in[14]; const float* h0 = (const float*)d_in[15];
    const float* g1 = (const float*)d_in[16]; const float* h1 = (const float*)d_in[17];
    const float* g2 = (const float*)d_in[18]; const float* h2 = (const float*)d_in[19];
    const float* wf = (const float*)d_in[20]; const float* bfv = (const float*)d_in[21];
    const float* binp = (const float*)d_in[22];

    // --- workspace arena (~165 MB) ---
    char* base = (char*)d_ws; size_t off = 0;
    auto alloc = [&](size_t bytes) -> void* {
        void* p = base + off; off = (off + bytes + 255) & ~(size_t)255; return p;
    };
    bf*  hcat0 = (bf*)alloc((size_t)ROWS * 512 * 2);
    bf*  hcat1 = (bf*)alloc((size_t)ROWS * 512 * 2);
    bf*  q0 = (bf*)alloc((size_t)ROWS * 256 * 2);
    bf*  k0 = (bf*)alloc((size_t)ROWS * 256 * 2);
    bf*  v0 = (bf*)alloc((size_t)ROWS * 256 * 2);
    bf*  q1 = (bf*)alloc((size_t)ROWS * 256 * 2);
    bf*  k1 = (bf*)alloc((size_t)ROWS * 256 * 2);
    bf*  v1 = (bf*)alloc((size_t)ROWS * 256 * 2);
    bf*  att0 = (bf*)alloc((size_t)ROWS * 256 * 2);     // reused later as P0
    bf*  att1 = (bf*)alloc((size_t)ROWS * 256 * 2);     // reused later as P1
    float* mrg0 = (float*)alloc((size_t)ROWS * 256 * 4);  // reused as delta0
    float* mrg1 = (float*)alloc((size_t)ROWS * 256 * 4);  // reused as delta1
    bf*  h10 = (bf*)alloc((size_t)ROWS * 512 * 2);
    bf*  h11 = (bf*)alloc((size_t)ROWS * 512 * 2);
    float* d20 = (float*)alloc((size_t)ROWS * 256 * 4);   // low half reused as dn0
    float* d21 = (float*)alloc((size_t)ROWS * 256 * 4);   // low half reused as dn1
    bf*  wqb = (bf*)alloc(65536 * 2);
    bf*  wkb = (bf*)alloc(65536 * 2);
    bf*  wvb = (bf*)alloc(65536 * 2);
    bf*  wmb = (bf*)alloc(65536 * 2);
    bf*  w1b = (bf*)alloc(262144 * 2);
    bf*  w2b = (bf*)alloc(131072 * 2);
    bf*  wfb = (bf*)alloc(65536 * 2);
    float* coup = (float*)alloc((size_t)BZ * SKN * SKN * 4);
    float* u = (float*)alloc((size_t)BZ * SKN * 4);
    float* v = (float*)alloc((size_t)BZ * SKN * 4);
    float* pm = (float*)alloc((size_t)BZ * CSLAB * SKN * 4);
    float* ps = (float*)alloc((size_t)BZ * CSLAB * SKN * 4);

    // --- weights -> bf16 ---
    f2bf_k<<<(65536 + 255) / 256, 256, 0, stream>>>(wq, wqb, 65536);
    f2bf_k<<<(65536 + 255) / 256, 256, 0, stream>>>(wk, wkb, 65536);
    f2bf_k<<<(65536 + 255) / 256, 256, 0, stream>>>(wv, wvb, 65536);
    f2bf_k<<<(65536 + 255) / 256, 256, 0, stream>>>(wm, wmb, 65536);
    f2bf_k<<<(262144 + 255) / 256, 256, 0, stream>>>(w1, w1b, 262144);
    f2bf_k<<<(131072 + 255) / 256, 256, 0, stream>>>(w2, w2b, 131072);
    f2bf_k<<<(65536 + 255) / 256, 256, 0, stream>>>(wf, wfb, 65536);

    // --- norm0 (transpose + LN) into first half of hcat ---
    trans_ln_k<<<dim3(NTOK / 32, BZ), 256, 0, stream>>>(desc0, g0, h0, hcat0, 512);
    trans_ln_k<<<dim3(NTOK / 32, BZ), 256, 0, stream>>>(desc1, g0, h0, hcat1, 512);

    const dim3 gP(256 / 64, ROWS / 128, 1);   // proj GEMMs: N=256, M=8192
    // q/k/v for both directions (bias epilogue, bf16 out)
    gemm_bf16_k<true, false, false, true><<<gP, 256, 0, stream>>>(
        hcat0, 512, 0, wqb, 256, 0, bq, nullptr, q0, 256, 0, 256, 1.f);
    gemm_bf16_k<true, false, false, true><<<gP, 256, 0, stream>>>(
        hcat0, 512, 0, wkb, 256, 0, bk, nullptr, k0, 256, 0, 256, 1.f);
    gemm_bf16_k<true, false, false, true><<<gP, 256, 0, stream>>>(
        hcat0, 512, 0, wvb, 256, 0, bv, nullptr, v0, 256, 0, 256, 1.f);
    gemm_bf16_k<true, false, false, true><<<gP, 256, 0, stream>>>(
        hcat1, 512, 0, wqb, 256, 0, bq, nullptr, q1, 256, 0, 256, 1.f);
    gemm_bf16_k<true, false, false, true><<<gP, 256, 0, stream>>>(
        hcat1, 512, 0, wkb, 256, 0, bk, nullptr, k1, 256, 0, 256, 1.f);
    gemm_bf16_k<true, false, false, true><<<gP, 256, 0, stream>>>(
        hcat1, 512, 0, wvb, 256, 0, bv, nullptr, v1, 256, 0, 256, 1.f);

    // --- cross attention (flash, WMMA) ---
    flash_attn_k<<<dim3(NTOK / 128, 4, BZ), 256, 0, stream>>>(q0, k1, v1, att0);
    flash_attn_k<<<dim3(NTOK / 128, 4, BZ), 256, 0, stream>>>(q1, k0, v0, att1);

    // --- merge GEMM (f32 out) + norm1 into second half of hcat ---
    gemm_bf16_k<true, false, true, false><<<gP, 256, 0, stream>>>(
        att0, 256, 0, wmb, 256, 0, bm, mrg0, nullptr, 256, 0, 256, 1.f);
    gemm_bf16_k<true, false, true, false><<<gP, 256, 0, stream>>>(
        att1, 256, 0, wmb, 256, 0, bm, mrg1, nullptr, 256, 0, 256, 1.f);
    row_ln_k<<<ROWS, 256, 0, stream>>>(mrg0, g1, h1, nullptr, hcat0 + 256, 512);
    row_ln_k<<<ROWS, 256, 0, stream>>>(mrg1, g1, h1, nullptr, hcat1 + 256, 512);

    // --- MLP: w1 (512x512) + exact GELU, then w2 (256x512) ---
    const dim3 gW1(512 / 64, ROWS / 128, 1);
    gemm_bf16_k<true, true, false, true><<<gW1, 256, 0, stream>>>(
        hcat0, 512, 0, w1b, 512, 0, c1, nullptr, h10, 512, 0, 512, 1.f);
    gemm_bf16_k<true, true, false, true><<<gW1, 256, 0, stream>>>(
        hcat1, 512, 0, w1b, 512, 0, c1, nullptr, h11, 512, 0, 512, 1.f);
    gemm_bf16_k<true, false, true, false><<<gP, 256, 0, stream>>>(
        h10, 512, 0, w2b, 512, 0, c2, d20, nullptr, 256, 0, 512, 1.f);
    gemm_bf16_k<true, false, true, false><<<gP, 256, 0, stream>>>(
        h11, 512, 0, w2b, 512, 0, c2, d21, nullptr, 256, 0, 512, 1.f);

    // --- norm2 -> delta (f32), then residual add with transpose -> bf16 ---
    row_ln_k<<<ROWS, 256, 0, stream>>>(d20, g2, h2, mrg0, nullptr, 0);
    row_ln_k<<<ROWS, 256, 0, stream>>>(d21, g2, h2, mrg1, nullptr, 0);
    bf* dn0 = (bf*)d20;   // reuse (d2 fully consumed)
    bf* dn1 = (bf*)d21;
    trans_add_k<<<dim3(NTOK / 32, BZ), 256, 0, stream>>>(desc0, mrg0, dn0);
    trans_add_k<<<dim3(NTOK / 32, BZ), 256, 0, stream>>>(desc1, mrg1, dn1);

    // --- final projection (reuse att buffers as P) ---
    bf* P0 = att0; bf* P1 = att1;
    gemm_bf16_k<true, false, false, true><<<gP, 256, 0, stream>>>(
        dn0, 256, 0, wfb, 256, 0, bfv, nullptr, P0, 256, 0, 256, 1.f);
    gemm_bf16_k<true, false, false, true><<<gP, 256, 0, stream>>>(
        dn1, 256, 0, wfb, 256, 0, bfv, nullptr, P1, 256, 0, 256, 1.f);

    // --- scores GEMM straight into coupling matrix (ldc = 2049, scale 1/16) ---
    gemm_bf16_k<false, false, true, false><<<dim3(NTOK / 64, NTOK / 128, BZ), 256, 0, stream>>>(
        P0, 256, (long long)NTOK * 256, P1, 256, (long long)NTOK * 256, nullptr,
        coup, nullptr, SKN, (long long)SKN * SKN, 256, 0.0625f);
    fill_bins_k<<<(BZ * SKN + 255) / 256, 256, 0, stream>>>(coup, binp);

    // --- Sinkhorn in log space (coupling fits in 192 MB L2) ---
    zero_k<<<(BZ * SKN + 255) / 256, 256, 0, stream>>>(u, BZ * SKN);
    zero_k<<<(BZ * SKN + 255) / 256, 256, 0, stream>>>(v, BZ * SKN);
    const int jblocks = (SKN + 255) / 256;
    for (int it = 0; it < 100; ++it) {
        sink_row_k<<<dim3(SKN, BZ), 256, 0, stream>>>(coup, v, u);
        sink_col_part_k<<<dim3(jblocks, BZ, CSLAB), 256, 0, stream>>>(coup, u, pm, ps);
        sink_col_comb_k<<<dim3(jblocks, BZ), 256, 0, stream>>>(pm, ps, v);
    }
    const size_t total = (size_t)BZ * SKN * SKN;
    sink_final_k<<<(unsigned)((total + 255) / 256), 256, 0, stream>>>(coup, u, v, (float*)d_out);
}
